// GNN_16982300688846
// MI455X (gfx1250) — compile-verified, matrix operand order checked by reading, not run on hardware
//
#include <hip/hip_runtime.h>
#include <hip/hip_fp16.h>
#include <math.h>

// ---------------------------------------------------------------------------
// GCN forward on MI455X (gfx1250, wave32).
//   roofline: message-passing scatter/gather dominates (~1.6 GB -> ~70us @23.3TB/s)
//   GEMMs (100000x64 @ 64x64) via v_wmma_f32_16x16x32_f16 (f32 accumulate),
//   B operand staged in LDS via global_load_async_to_lds_b128 (ASYNCcnt path).
// ---------------------------------------------------------------------------

typedef __attribute__((ext_vector_type(16))) _Float16 v16h;
typedef __attribute__((ext_vector_type(8)))  _Float16 v8h;
typedef __attribute__((ext_vector_type(8)))  float    v8f;
typedef __attribute__((ext_vector_type(4)))  int      v4i;

#define N_NODES   100000
#define N_EDGES   1600000
#define FDIM      64
#define NGRAPH    64
#define NOUT      8

#if defined(__has_builtin)
#  if __has_builtin(__builtin_amdgcn_global_load_async_to_lds_b128) && \
      __has_builtin(__builtin_amdgcn_s_wait_asynccnt)
#    define USE_ASYNC_LDS 1
#  endif
#endif
#ifndef USE_ASYNC_LDS
#  define USE_ASYNC_LDS 0
#endif

union AF { v16h v; v8h h[2]; };

// ---------------------------------------------------------------------------
// init: deg = 1.0 (self loop), pooled = 0, cnt = 0
// ---------------------------------------------------------------------------
__global__ __launch_bounds__(256)
void k_init(float* __restrict__ deg, float* __restrict__ pooled, float* __restrict__ cnt) {
    int i = blockIdx.x * blockDim.x + threadIdx.x;
    if (i < N_NODES) deg[i] = 1.0f;
    if (i < NGRAPH * FDIM) pooled[i] = 0.0f;
    if (i < NGRAPH) cnt[i] = 0.0f;
}

__global__ __launch_bounds__(256)
void k_degree(const int* __restrict__ ei, float* __restrict__ deg) {
    int e = blockIdx.x * blockDim.x + threadIdx.x;
    if (e < N_EDGES) atomicAdd(&deg[ei[N_EDGES + e]], 1.0f);
}

__global__ __launch_bounds__(256)
void k_dinv(float* __restrict__ dinv) {
    int i = blockIdx.x * blockDim.x + threadIdx.x;
    if (i < N_NODES) dinv[i] = rsqrtf(dinv[i]);
}

__global__ __launch_bounds__(256)
void k_cnt(const int* __restrict__ batch, float* __restrict__ cnt) {
    int i = blockIdx.x * blockDim.x + threadIdx.x;
    if (i < N_NODES) atomicAdd(&cnt[batch[i]], 1.0f);
}

// x (f32) -> f16 staging buffer for WMMA A operand
__global__ __launch_bounds__(256)
void k_prep_x(const float* __restrict__ x, _Float16* __restrict__ xh) {
    size_t i = (size_t)blockIdx.x * blockDim.x + threadIdx.x;
    if (i < (size_t)N_NODES * FDIM) xh[i] = (_Float16)x[i];
}

// W1/W2 (f32, KxN row-major) -> f16 transposed (NxK) so B frags are contiguous loads
__global__ __launch_bounds__(256)
void k_prep_w(const float* __restrict__ W1, const float* __restrict__ W2,
              _Float16* __restrict__ Wt1, _Float16* __restrict__ Wt2) {
    int i = blockIdx.x * blockDim.x + threadIdx.x;
    if (i >= 2 * FDIM * FDIM) return;
    int which = i >> 12;
    int idx = i & 4095;
    int k = idx >> 6, n = idx & 63;
    const float* W = which ? W2 : W1;
    _Float16* Wt = which ? Wt2 : Wt1;
    Wt[n * FDIM + k] = (_Float16)W[k * FDIM + n];
}

// ---------------------------------------------------------------------------
// WMMA GEMM: hw = X(f16) @ W (via transposed f16 Wt), also seed
// agg = hw * dinv[row]^2  (self-loop contribution, scatter accumulates on top).
// 128 threads = 4 waves; each wave computes one 16-row stripe x 64 cols.
// Wt (8KB) staged into LDS once per block with async-to-LDS copies; B frags
// then come from ds_load_b128 so v_wmma is not gated on global loadcnt.
// ---------------------------------------------------------------------------
__global__ __launch_bounds__(128)
void k_gemm(const _Float16* __restrict__ X, const _Float16* __restrict__ Wt,
            const float* __restrict__ dinv, float* __restrict__ hw,
            float* __restrict__ agg) {
    __shared__ _Float16 sW[FDIM * FDIM];   // 8KB, whole transposed weight

    const int t    = threadIdx.x;
    const int wave = t >> 5;
    const int lane = t & 31;
    const int half = lane >> 4;      // K-group select per ISA A-layout
    const int lr   = lane & 15;      // row (A) / col (B) within tile

    // ---- cooperative stage of Wt -> LDS (each thread moves 4 x 16B) ----
#if USE_ASYNC_LDS
#pragma unroll
    for (int c = 0; c < 4; ++c) {
        const int idx = (t + c * 128) * 8;   // 8 halves = 16 bytes
        __builtin_amdgcn_global_load_async_to_lds_b128(
            (__attribute__((address_space(1))) v4i*)(Wt + idx),
            (__attribute__((address_space(3))) v4i*)(sW + idx),
            0, 0);
    }
    __builtin_amdgcn_s_wait_asynccnt(0);
#else
#pragma unroll
    for (int c = 0; c < 4; ++c) {
        const int idx = (t + c * 128) * 8;
        *(v8h*)(sW + idx) = *(const v8h*)(Wt + idx);
    }
#endif
    __syncthreads();

    const int m0 = (blockIdx.x * 4 + wave) * 16;
    if (m0 >= N_NODES) return;       // wave-uniform: EXEC stays all-ones for WMMA

    // A fragments: 16x32 f16, lane<16 holds K {kb..kb+7, kb+16..kb+23}, lane>=16 shifted by 8
    const _Float16* arow = X + (size_t)(m0 + lr) * FDIM;
    const int ko = half * 8;
    AF a0, a1;
    a0.h[0] = *(const v8h*)(arow + ko);
    a0.h[1] = *(const v8h*)(arow + ko + 16);
    a1.h[0] = *(const v8h*)(arow + 32 + ko);
    a1.h[1] = *(const v8h*)(arow + 32 + ko + 16);

    v8f acc[4] = {};
#pragma unroll
    for (int n = 0; n < 4; ++n) {
        const _Float16* brow = sW + (n * 16 + lr) * FDIM;   // LDS: ds_load_b128
        AF b0, b1;
        b0.h[0] = *(const v8h*)(brow + ko);
        b0.h[1] = *(const v8h*)(brow + ko + 16);
        b1.h[0] = *(const v8h*)(brow + 32 + ko);
        b1.h[1] = *(const v8h*)(brow + 32 + ko + 16);
        acc[n] = __builtin_amdgcn_wmma_f32_16x16x32_f16(
            false, a0.v, false, b0.v, (short)0, acc[n], false, false);
        acc[n] = __builtin_amdgcn_wmma_f32_16x16x32_f16(
            false, a1.v, false, b1.v, (short)0, acc[n], false, false);
    }

    // dinv^2 for the 8 rows this lane writes (C layout: VGPR r -> M = r + 8*half)
    float dsq[8];
#pragma unroll
    for (int r = 0; r < 8; ++r) {
        float dv = dinv[m0 + half * 8 + r];
        dsq[r] = dv * dv;
    }

#pragma unroll
    for (int n = 0; n < 4; ++n) {
#pragma unroll
        for (int r = 0; r < 8; ++r) {
            size_t row = (size_t)(m0 + half * 8 + r);
            int colx = n * 16 + lr;
            float v = acc[n][r];
            hw[row * FDIM + colx] = v;
            agg[row * FDIM + colx] = v * dsq[r];
        }
    }
}

// ---------------------------------------------------------------------------
// Edge scatter: agg[col] += hw[row] * dinv[row]*dinv[col].
// 16 lanes per edge, float4 per lane; prefetch next edge's source row.
// ---------------------------------------------------------------------------
__global__ __launch_bounds__(256)
void k_scatter(const float* __restrict__ hw, const int* __restrict__ ei,
               const float* __restrict__ dinv, float* __restrict__ agg) {
    const int li = threadIdx.x & 15;
    long long group = ((long long)blockIdx.x * blockDim.x + threadIdx.x) >> 4;
    const long long ngroups = ((long long)gridDim.x * blockDim.x) >> 4;
    for (long long e = group; e < N_EDGES; e += ngroups) {
        long long en = e + ngroups;
        if (en < N_EDGES) {
            int rn = ei[en];
            __builtin_prefetch(hw + (size_t)rn * FDIM + li * 4, 0, 3);
        }
        int row = ei[e];
        int col = ei[N_EDGES + e];
        float norm = dinv[row] * dinv[col];
        const float4 v = *(const float4*)(hw + (size_t)row * FDIM + li * 4);
        float* dst = agg + (size_t)col * FDIM + li * 4;
        atomicAdd(dst + 0, v.x * norm);
        atomicAdd(dst + 1, v.y * norm);
        atomicAdd(dst + 2, v.z * norm);
        atomicAdd(dst + 3, v.w * norm);
    }
}

// combine layer-1: xh = f16(relu(agg + b))  (feeds layer-2 WMMA directly)
__global__ __launch_bounds__(256)
void k_combine_f16(const float* __restrict__ agg, const float* __restrict__ b,
                   _Float16* __restrict__ xh) {
    size_t i = (size_t)blockIdx.x * blockDim.x + threadIdx.x;
    if (i < (size_t)N_NODES * FDIM) {
        int j = (int)(i & 63);
        xh[i] = (_Float16)fmaxf(agg[i] + b[j], 0.0f);
    }
}

// combine layer-2 fused with mean-pool numerator: pooled[batch[n]] += relu(agg + b)
__global__ __launch_bounds__(256)
void k_combine_pool(const float* __restrict__ agg, const float* __restrict__ b,
                    const int* __restrict__ batch, float* __restrict__ pooled) {
    size_t i = (size_t)blockIdx.x * blockDim.x + threadIdx.x;
    if (i < (size_t)N_NODES * FDIM) {
        int node = (int)(i >> 6);
        int j = (int)(i & 63);
        float v = fmaxf(agg[i] + b[j], 0.0f);
        atomicAdd(&pooled[(size_t)batch[node] * FDIM + j], v);
    }
}

// ---------------------------------------------------------------------------
// Head MLP + log_softmax, single block (64x64 @ 64x64, 64x64 @ 64x8 — tiny)
// ---------------------------------------------------------------------------
__global__ __launch_bounds__(512)
void k_head(const float* __restrict__ pooled, const float* __restrict__ cnt,
            const float* __restrict__ Wf1, const float* __restrict__ bf1,
            const float* __restrict__ Wf2, const float* __restrict__ bf2,
            float* __restrict__ out) {
    __shared__ float sp[NGRAPH * FDIM];
    __shared__ float sz[NGRAPH * FDIM];
    __shared__ float so[NGRAPH * NOUT];
    const int tid = threadIdx.x;

    for (int i = tid; i < NGRAPH * FDIM; i += 512) {
        int g = i >> 6;
        sp[i] = pooled[i] / fmaxf(cnt[g], 1.0f);
    }
    __syncthreads();
    for (int i = tid; i < NGRAPH * FDIM; i += 512) {
        int g = i >> 6, j = i & 63;
        float s = bf1[j];
#pragma unroll 8
        for (int k = 0; k < FDIM; ++k) s += sp[g * FDIM + k] * Wf1[k * FDIM + j];
        sz[i] = fmaxf(s, 0.0f);
    }
    __syncthreads();
    for (int i = tid; i < NGRAPH * NOUT; i += 512) {
        int g = i >> 3, o = i & 7;
        float s = bf2[o];
#pragma unroll 8
        for (int k = 0; k < FDIM; ++k) s += sz[g * FDIM + k] * Wf2[k * NOUT + o];
        so[i] = s;
    }
    __syncthreads();
    if (tid < NGRAPH) {
        float m = -1e30f;
        for (int o = 0; o < NOUT; ++o) m = fmaxf(m, so[tid * NOUT + o]);
        float s = 0.0f;
        for (int o = 0; o < NOUT; ++o) s += expf(so[tid * NOUT + o] - m);
        float lse = m + logf(s);
        for (int o = 0; o < NOUT; ++o) out[tid * NOUT + o] = so[tid * NOUT + o] - lse;
    }
}

// ---------------------------------------------------------------------------
extern "C" void kernel_launch(void* const* d_in, const int* in_sizes, int n_in,
                              void* d_out, int out_size, void* d_ws, size_t ws_size,
                              hipStream_t stream) {
    (void)in_sizes; (void)n_in; (void)out_size; (void)ws_size;

    const float* x     = (const float*)d_in[0];
    const int*   ei    = (const int*)d_in[1];   // [2, E] int32
    const int*   batch = (const int*)d_in[2];
    const float* W1    = (const float*)d_in[3];
    const float* b1    = (const float*)d_in[4];
    const float* W2    = (const float*)d_in[5];
    const float* b2    = (const float*)d_in[6];
    const float* Wf1   = (const float*)d_in[7];
    const float* bf1   = (const float*)d_in[8];
    const float* Wf2   = (const float*)d_in[9];
    const float* bf2   = (const float*)d_in[10];
    float* out = (float*)d_out;

    // workspace carve-up (~64.5 MB total), 256B-aligned slices
    size_t off = 0;
    auto alloc = [&](size_t bytes) -> void* {
        void* p = (char*)d_ws + off;
        off += (bytes + 255) & ~(size_t)255;
        return p;
    };
    float*    dinv   = (float*)alloc((size_t)N_NODES * 4);               // deg then rsqrt in-place
    _Float16* xh     = (_Float16*)alloc((size_t)N_NODES * FDIM * 2);     // f16 layer input
    float*    hw     = (float*)alloc((size_t)N_NODES * FDIM * 4);        // h = X @ W
    float*    agg    = (float*)alloc((size_t)N_NODES * FDIM * 4);        // aggregated messages
    _Float16* Wt1    = (_Float16*)alloc((size_t)FDIM * FDIM * 2);
    _Float16* Wt2    = (_Float16*)alloc((size_t)FDIM * FDIM * 2);
    float*    pooled = (float*)alloc((size_t)NGRAPH * FDIM * 4);
    float*    cnt    = (float*)alloc((size_t)NGRAPH * 4);

    const int T = 256;
    const int gN   = (N_NODES + T - 1) / T;
    const int gE   = (N_EDGES + T - 1) / T;
    const int gNF  = (int)(((size_t)N_NODES * FDIM + T - 1) / T);   // 25000
    const int gGemm = (N_NODES / 16 + 3) / 4;                       // 1563 blocks of 4 waves
    const int gScat = 50000;                                        // 2 edges per 16-lane group

    k_init  <<<gN,  T, 0, stream>>>(dinv, pooled, cnt);
    k_degree<<<gE,  T, 0, stream>>>(ei, dinv);
    k_dinv  <<<gN,  T, 0, stream>>>(dinv);
    k_cnt   <<<gN,  T, 0, stream>>>(batch, cnt);
    k_prep_x<<<gNF, T, 0, stream>>>(x, xh);
    k_prep_w<<<(2 * FDIM * FDIM + T - 1) / T, T, 0, stream>>>(W1, W2, Wt1, Wt2);

    // layer 1
    k_gemm       <<<gGemm, 128, 0, stream>>>(xh, Wt1, dinv, hw, agg);
    k_scatter    <<<gScat, T,   0, stream>>>(hw, ei, dinv, agg);
    k_combine_f16<<<gNF,   T,   0, stream>>>(agg, b1, xh);

    // layer 2
    k_gemm        <<<gGemm, 128, 0, stream>>>(xh, Wt2, dinv, hw, agg);
    k_scatter     <<<gScat, T,   0, stream>>>(hw, ei, dinv, agg);
    k_combine_pool<<<gNF,   T,   0, stream>>>(agg, b2, batch, pooled);

    // head
    k_head<<<1, 512, 0, stream>>>(pooled, cnt, Wf1, bf1, Wf2, bf2, out);
}
